// TripletLoss_36129264894687
// MI455X (gfx1250) — matrix operand order repaired and checked by reading, not compiled, not run on hardware
//
#include <hip/hip_runtime.h>
#include <hip/hip_bf16.h>

typedef _Float16 h16;
typedef __attribute__((ext_vector_type(16))) _Float16 v16h;
typedef __attribute__((ext_vector_type(8)))  _Float16 v8h;
typedef __attribute__((ext_vector_type(4)))  _Float16 v4h;
typedef __attribute__((ext_vector_type(8)))  float    v8f;

constexpr int   kN      = 512;
constexpr int   kD      = 128;
constexpr float kMargin = 0.2f;

// ---------------------------------------------------------------------------
// Kernel 0: exact f32 row norms + f32 -> f16 copy of z (for WMMA loads).
// ---------------------------------------------------------------------------
__global__ void tl_prep(const float* __restrict__ z,
                        h16* __restrict__ zh,
                        float* __restrict__ norms) {
  int row = blockIdx.x * blockDim.x + threadIdx.x;
  if (row >= kN) return;
  const float4* zr = (const float4*)(z + row * kD);
  v4h* zo = (v4h*)(zh + row * kD);
  float s = 0.f;
#pragma unroll
  for (int k = 0; k < kD / 4; ++k) {
    float4 v = zr[k];
    s += v.x * v.x + v.y * v.y + v.z * v.z + v.w * v.w;
    v4h h;
    h[0] = (h16)v.x; h[1] = (h16)v.y; h[2] = (h16)v.z; h[3] = (h16)v.w;
    zo[k] = h;
  }
  norms[row] = s;
}

// ---------------------------------------------------------------------------
// Kernel 1: Gram matrix via v_wmma_f32_16x16x32_f16, one wave per 16x16 tile,
// epilogue converts to Euclidean distances: d = sqrt(max(ni + nj - 2*G, 0)).
//
// A fragment (16x32 f16): lane<16 -> row M=lane, K = kb+{0..15};
// lane>=16 -> row M=lane-16, K = kb+16+{0..15}.
// B fragment (32x16): lane<16 -> col N=lane, K = kb+{0..15}; lane>=16 -> col
// N=lane-16, K = kb+16+{0..15}. With B = z^T both are contiguous row loads.
// ---------------------------------------------------------------------------
__global__ void tl_gram_dist(const h16* __restrict__ zh,
                             const float* __restrict__ norms,
                             float* __restrict__ dmat) {
  const int lane = threadIdx.x;          // 0..31, one wave per block
  const int I = blockIdx.y * 16;
  const int J = blockIdx.x * 16;
  const int m  = lane & 15;
  const int hi = lane >> 4;              // 0 or 1 (lane group)

  v8f acc = {};
#pragma unroll
  for (int kb = 0; kb < kD; kb += 32) {
    // A: row I+m, K = kb + 16*hi + {0..15}
    const h16* arow = zh + (I + m) * kD + kb + 16 * hi;
    v8h alo = *(const v8h*)(arow);
    v8h ahi = *(const v8h*)(arow + 8);
    v16h a;
#pragma unroll
    for (int j = 0; j < 8; ++j) { a[j] = alo[j]; a[8 + j] = ahi[j]; }

    // B: column J+m of z^T == row J+m of z, K = kb + 16*hi + {0..15}
    const h16* brow = zh + (J + m) * kD + kb + 16 * hi;
    v16h b = *(const v16h*)(brow);

    // (neg_a, A, neg_b, B, c_mod, C, reuse_a, reuse_b)
    acc = __builtin_amdgcn_wmma_f32_16x16x32_f16(false, a, false, b,
                                                 (short)0, acc, false, false);
  }

  // C/D layout: VGPR r, lane<16 -> M = r, lane>=16 -> M = r+8; N = lane&15.
#pragma unroll
  for (int r = 0; r < 8; ++r) {
    int row = I + r + 8 * hi;
    int col = J + m;
    float sq = norms[row] + norms[col] - 2.0f * acc[r];
    float dv = (sq > 0.f) ? sqrtf(sq) : 0.f;
    dmat[row * kN + col] = dv;
  }
}

// ---------------------------------------------------------------------------
// Kernel 2: one 256-thread block per anchor. Stage d[a,:] + labels in LDS,
// deterministically compact positive/negative distance lists, then stride the
// npos*nneg pair space. Fixed-order partial sums + LDS tree reduction keep the
// result bitwise deterministic (no float atomics).
// ---------------------------------------------------------------------------
__global__ void tl_triplet(const float* __restrict__ dmat,
                           const int* __restrict__ labels,
                           float* __restrict__ psum,
                           int* __restrict__ peg,
                           int* __restrict__ ptrip) {
  __shared__ float s_drow[kN];
  __shared__ int   s_lab[kN];
  __shared__ float s_dpos[kN];
  __shared__ float s_dneg[kN];
  __shared__ int   s_cnt[2];
  __shared__ float s_rf[256];
  __shared__ int   s_ri[256];

  const int a   = blockIdx.x;
  const int tid = threadIdx.x;

  for (int j = tid; j < kN; j += 256) {
    s_drow[j] = dmat[a * kN + j];
    s_lab[j]  = labels[j];
  }
  __syncthreads();

  if (tid == 0) {                         // deterministic compaction (j order)
    const int la = s_lab[a];
    int np = 0, nn = 0;
    for (int j = 0; j < kN; ++j) {
      if (j == a) continue;
      float dv = s_drow[j];
      if (s_lab[j] == la) s_dpos[np++] = dv;
      else                s_dneg[nn++] = dv;
    }
    s_cnt[0] = np;
    s_cnt[1] = nn;
  }
  __syncthreads();

  const int np    = s_cnt[0];
  const int nn    = s_cnt[1];
  const int total = np * nn;

  float lsum = 0.f;
  int   leg  = 0;
  for (int idx = tid; idx < total; idx += 256) {
    int p = idx / nn;
    int q = idx - p * nn;
    float l = s_dpos[p] - s_dneg[q] + kMargin;
    if (l > 0.f) { lsum += l; leg += 1; }
  }

  s_rf[tid] = lsum;
  s_ri[tid] = leg;
  __syncthreads();
  for (int s = 128; s > 0; s >>= 1) {
    if (tid < s) { s_rf[tid] += s_rf[tid + s]; s_ri[tid] += s_ri[tid + s]; }
    __syncthreads();
  }
  if (tid == 0) {
    psum[a]  = s_rf[0];
    peg[a]   = s_ri[0];
    ptrip[a] = total;
  }
}

// ---------------------------------------------------------------------------
// Kernel 3: reduce 512 per-anchor partials; out[0]=loss, out[1]=n_eg (float).
// ---------------------------------------------------------------------------
__global__ void tl_final(const float* __restrict__ psum,
                         const int* __restrict__ peg,
                         const int* __restrict__ ptrip,
                         float* __restrict__ out) {
  __shared__ float rf[256];
  __shared__ int   re[256];
  __shared__ int   rt[256];
  const int tid = threadIdx.x;
  rf[tid] = psum[tid] + psum[tid + 256];
  re[tid] = peg[tid]  + peg[tid + 256];
  rt[tid] = ptrip[tid] + ptrip[tid + 256];
  __syncthreads();
  for (int s = 128; s > 0; s >>= 1) {
    if (tid < s) {
      rf[tid] += rf[tid + s];
      re[tid] += re[tid + s];
      rt[tid] += rt[tid + s];
    }
    __syncthreads();
  }
  if (tid == 0) {
    float nt = (float)rt[0];
    out[0] = (nt > 0.f) ? (rf[0] / nt) : 0.f;
    out[1] = (float)re[0];
  }
}

// ---------------------------------------------------------------------------
// Launcher. Workspace layout (bytes):
//   [0,            131072)  f16 copy of z         (512*128*2)
//   [131072,       133120)  f32 row norms         (512*4)
//   [133120,      1181696)  f32 distance matrix   (512*512*4)
//   [1181696,     1183744)  f32 per-anchor loss partials
//   [1183744,     1185792)  i32 per-anchor n_eg partials
//   [1185792,     1187840)  i32 per-anchor n_triplet partials
// ---------------------------------------------------------------------------
extern "C" void kernel_launch(void* const* d_in, const int* in_sizes, int n_in,
                              void* d_out, int out_size, void* d_ws, size_t ws_size,
                              hipStream_t stream) {
  const float* z      = (const float*)d_in[0];
  const int*   labels = (const int*)d_in[1];
  float*       out    = (float*)d_out;

  char*  ws    = (char*)d_ws;
  h16*   zh    = (h16*)(ws);
  float* norms = (float*)(ws + 131072);
  float* dmat  = (float*)(ws + 133120);
  float* psum  = (float*)(ws + 1181696);
  int*   peg   = (int*)(ws + 1183744);
  int*   ptrip = (int*)(ws + 1185792);

  tl_prep<<<dim3(2), dim3(256), 0, stream>>>(z, zh, norms);
  tl_gram_dist<<<dim3(kN / 16, kN / 16), dim3(32), 0, stream>>>(zh, norms, dmat);
  tl_triplet<<<dim3(kN), dim3(256), 0, stream>>>(dmat, labels, psum, peg, ptrip);
  tl_final<<<dim3(1), dim3(256), 0, stream>>>(psum, peg, ptrip, out);
}
